// AttentionDecoder2D_16106127360045
// MI455X (gfx1250) — matrix operand order repaired and checked by reading, not compiled, not run on hardware
//
#include <hip/hip_runtime.h>
#include <hip/hip_bf16.h>

// ---------------- problem constants (from reference) ----------------
constexpr int B  = 128;     // batch
constexpr int T  = 20;      // time steps
constexpr int V  = 10000;   // vocab
constexpr int H  = 512;     // hidden
constexpr int F  = 49;      // attention locations
constexpr int H2 = 2 * H;   // 1024
constexpr int H4 = 4 * H;   // 2048

// ---------------- vector types ----------------
typedef __attribute__((ext_vector_type(2)))  float  v2f;
typedef __attribute__((ext_vector_type(8)))  float  v8f;
typedef __attribute__((ext_vector_type(16))) __bf16 v16bf;
typedef __attribute__((ext_vector_type(8)))  __bf16 bf16x8;
typedef __attribute__((ext_vector_type(4)))  unsigned int u32x4;
typedef __attribute__((ext_vector_type(4)))  int    i32x4;
typedef __attribute__((ext_vector_type(8)))  int    i32x8;

__device__ __forceinline__ float sigf(float x) { return 1.0f / (1.0f + expf(-x)); }

#if __has_builtin(__builtin_amdgcn_tensor_load_to_lds) && __has_builtin(__builtin_amdgcn_s_wait_tensorcnt)
#define USE_TDM 1
#else
#define USE_TDM 0
#endif

// ================= A-operand accessors for the f32 WMMA GEMM =================
struct ARow {           // plain row-major A
  const float* p; int ld;
  __device__ float operator()(int m, int k) const { return p[(size_t)m * ld + k]; }
};
struct AEmb {           // gathered embedding rows: m = t*B + b, tok = cap[b*T + t]
  const float* emb; const int* cap;
  __device__ float operator()(int m, int k) const {
    int b = m & (B - 1); int t = m >> 7;            // B == 128
    int tok = cap[b * T + t];
    return emb[(size_t)tok * H + k];
  }
};
struct AArea {          // feat[b][f][k] = area[b][k][f] (area is [B,H,F]); m = b*F + f
  const float* area;
  __device__ float operator()(int m, int k) const {
    int b = m / F, f = m - b * F;
    return area[((size_t)b * H + k) * F + f];
  }
};

// ================= generic f32 WMMA GEMM (16x16 tile per wave) =================
// C[M,N] = A @ B (+add0 +add1 +colBias).  K multiple of 4, M,N multiples of 16.
// V_WMMA_F32_16X16X4_F32 layout: A lane=M row, (half,vgpr)=K: v0={K0,K2}, v1={K1,K3}.
// B mirrored (lane=N col). C/D: vgpr r -> row m0+r+8*half, col n0+lane%16.
template <typename AG>
__global__ void __launch_bounds__(256)
gemm_f32_wmma(AG ag, const float* __restrict__ Bp, int ldb,
              float* __restrict__ C, int ldc, int Ntiles, int Tiles, int K,
              const float* __restrict__ add0, const float* __restrict__ add1,
              const float* __restrict__ colBias)
{
  int wave = (blockIdx.x * blockDim.x + threadIdx.x) >> 5;
  if (wave >= Tiles) return;                    // wave-uniform: EXEC stays all-ones
  int lane = threadIdx.x & 31;
  int half = lane >> 4;
  int lm   = lane & 15;
  int tn = wave % Ntiles, tm = wave / Ntiles;
  int m0 = tm * 16, n0 = tn * 16;
  int row = m0 + lm, col = n0 + lm;

  v8f acc = {};
  for (int k = 0; k < K; k += 4) {
    int ka = k + 2 * half;
    v2f a, b;
    a.x = ag(row, ka);
    a.y = ag(row, ka + 1);
    b.x = Bp[(size_t)ka * ldb + col];
    b.y = Bp[(size_t)(ka + 1) * ldb + col];
    acc = __builtin_amdgcn_wmma_f32_16x16x4_f32(false, a, false, b,
                                                (short)0, acc, false, false);
  }
#pragma unroll
  for (int r = 0; r < 8; ++r) {
    int rr = m0 + r + 8 * half;
    size_t idx = (size_t)rr * ldc + col;
    float v = acc[r];
    if (add0)    v += add0[idx];
    if (add1)    v += add1[idx];
    if (colBias) v += colBias[col];
    C[idx] = v;
  }
}

// ================= BF16 output projection: out = Hcat @ W_out^T(+b) =================
// M = T*B = 2560 rows (bf16 [M,1024]), N = V = 10000 cols via Bt = W_out^T bf16 [V,1024].
//
// Blocking: wave = 32x80 super-tile (2 M-tiles x 5 N-tiles, 10 f32 accumulators).
// Workgroup = 8 waves stacked in M (256 rows) sharing ONE 80-column B slab.
// B slab is staged LDS-double-buffered in K-chunks of 128 (21.25 KB each incl. pad),
// via the Tensor Data Mover when available (pad_enable makes HW produce the padded
// pitch), else cooperative 16B copies.  Grid = (2560/256) * (10000/80) = 1250 full WGs.
constexpr int KC      = 128;                 // K chunk
constexpr int NCHUNK  = H2 / KC;             // 8
constexpr int NSLAB   = 80;                  // columns per WG  (5 N-tiles)
constexpr int BPITCH  = KC + 8;              // 136 bf16 = 272 B row pitch (bank spread)
constexpr int BUFELEM = NSLAB * BPITCH;      // 10880 bf16 per buffer

#if USE_TDM
// Build D# per CDNA5 ISA and issue TENSOR_LOAD_TO_LDS.
// Tile: 128 elems x 80 rows of 2B data, row stride 1024 elems, LDS pad 4 dwords
// after every 64 dwords (-> 272B pitch).
__device__ __forceinline__ void tdm_load_slab(unsigned lds_addr, const __bf16* gptr)
{
  unsigned long long ga = (unsigned long long)(uintptr_t)gptr;
  u32x4 g0;
  g0[0] = 1u;                                             // count=1 (valid descriptor)
  g0[1] = lds_addr;                                       // lds_addr[31:0]
  g0[2] = (unsigned)(ga & 0xffffffffu);                   // global_addr[31:0]
  g0[3] = (unsigned)((ga >> 32) & 0x01ffffffu)            // global_addr[56:32]
        | (2u << 30);                                     // type = 2 ("image")
  i32x8 g1;
  g1[0] = (1 << 16)    // data_size = 1 -> 2 bytes
        | (1 << 20)    // pad_enable
        | (5 << 22)    // pad_interval = 64 dwords (256B)
        | (3 << 25);   // pad_amount   = 4 dwords (16B)
  g1[1] = (int)((unsigned)KC << 16);        // tensor_dim0 = 128   (bits 79:48)
  g1[2] = (int)((unsigned)NSLAB << 16);     // tensor_dim1 = 80    (bits 111:80)
  g1[3] = (int)((unsigned)KC << 16);        // tile_dim0  = 128    (bits 127:112)
  g1[4] = NSLAB;                            // tile_dim1  = 80, tile_dim2 = 0
  g1[5] = H2;                               // tensor_dim0_stride = 1024
  g1[6] = 0;
  g1[7] = 0;
  i32x4 z4 = {0, 0, 0, 0};
#if __clang_major__ >= 23
  i32x8 z8 = {0, 0, 0, 0, 0, 0, 0, 0};
  __builtin_amdgcn_tensor_load_to_lds(g0, g1, z4, z4, z8, 0);
#else
  __builtin_amdgcn_tensor_load_to_lds(g0, g1, z4, z4, 0);
#endif
}
#endif

__global__ void __launch_bounds__(256)
outproj_bf16_wmma(const __bf16* __restrict__ A, const __bf16* __restrict__ Bt,
                  const float* __restrict__ bias, float* __restrict__ out)
{
  __shared__ __bf16 Bs[2 * BUFELEM];

  int tid  = threadIdx.x;
  int lane = tid & 31, half = lane >> 4, lm = lane & 15;
  int wv   = tid >> 5;                                    // wave 0..7
  int ns   = blockIdx.x % 125;                            // N super-tile
  int ms   = blockIdx.x / 125;                            // M block (256 rows)
  int n0g  = ns * NSLAB;
  int wgM  = ms * 256 + wv * 32;                          // this wave's first row

  const __bf16* arow0 = A + (size_t)(wgM + lm) * H2;      // M-tile 0 row
  const __bf16* arow1 = arow0 + (size_t)16 * H2;          // M-tile 1 row
  const __bf16* bslab = Bt + (size_t)n0g * H2;            // 80-row B slab base

  v8f acc[2][5];
#pragma unroll
  for (int i = 0; i < 2; ++i)
#pragma unroll
    for (int j = 0; j < 5; ++j) acc[i][j] = (v8f){};

#if USE_TDM
  if (wv == 0) {
    tdm_load_slab((unsigned)(uintptr_t)&Bs[0], bslab);    // chunk 0 -> buf 0
    __builtin_amdgcn_s_wait_tensorcnt(0);
  }
  __syncthreads();
#else
  // cooperative stage of chunk 0: 80 rows x 256B, pitch 272B
  for (int i = tid; i < NSLAB * (KC / 8); i += 256) {     // 1280 x 16B
    int r = i >> 4, cc = i & 15;
    *(bf16x8*)&Bs[r * BPITCH + cc * 8] =
        *(const bf16x8*)(bslab + (size_t)r * H2 + cc * 8);
  }
  __syncthreads();
#endif

  for (int c = 0; c < NCHUNK; ++c) {
    int buf = c & 1;
    const __bf16* bbase = &Bs[buf * BUFELEM];

#if USE_TDM
    if (c + 1 < NCHUNK && wv == 0)
      tdm_load_slab((unsigned)(uintptr_t)&Bs[(~c & 1) * BUFELEM],
                    bslab + (c + 1) * KC);
#else
    if (c + 1 < NCHUNK) {
      const __bf16* src = bslab + (c + 1) * KC;
      __bf16* dst = &Bs[(~c & 1) * BUFELEM];
      for (int i = tid; i < NSLAB * (KC / 8); i += 256) {
        int r = i >> 4, cc = i & 15;
        *(bf16x8*)&dst[r * BPITCH + cc * 8] =
            *(const bf16x8*)(src + (size_t)r * H2 + cc * 8);
      }
    }
#endif

    // ---- compute this chunk: 4 k-steps of 32, 10 WMMAs each ----
#pragma unroll
    for (int ks = 0; ks < KC; ks += 32) {
      int kg = c * KC + ks;
      bf16x8 a0l = *(const bf16x8*)(arow0 + kg + 8 * half);
      bf16x8 a0h = *(const bf16x8*)(arow0 + kg + 16 + 8 * half);
      bf16x8 a1l = *(const bf16x8*)(arow1 + kg + 8 * half);
      bf16x8 a1h = *(const bf16x8*)(arow1 + kg + 16 + 8 * half);
      v16bf a0, a1;
#pragma unroll
      for (int i = 0; i < 8; ++i) {
        a0[i] = a0l[i]; a0[8 + i] = a0h[i];
        a1[i] = a1l[i]; a1[8 + i] = a1h[i];
      }
#pragma unroll
      for (int j = 0; j < 5; ++j) {
        const __bf16* bp = bbase + (j * 16 + lm) * BPITCH + ks + 8 * half;
        bf16x8 bl = *(const bf16x8*)bp;
        bf16x8 bh = *(const bf16x8*)(bp + 16);
        v16bf bv;
#pragma unroll
        for (int i = 0; i < 8; ++i) { bv[i] = bl[i]; bv[8 + i] = bh[i]; }
        acc[0][j] = __builtin_amdgcn_wmma_f32_16x16x32_bf16(false, a0, false, bv,
                                                            (short)0, acc[0][j], false, false);
        acc[1][j] = __builtin_amdgcn_wmma_f32_16x16x32_bf16(false, a1, false, bv,
                                                            (short)0, acc[1][j], false, false);
      }
    }

#if USE_TDM
    if (c + 1 < NCHUNK && wv == 0) __builtin_amdgcn_s_wait_tensorcnt(0);
#endif
    __syncthreads();   // release buf for overwrite / publish next chunk
  }

  // ---- epilogue: out is [B,T,V], row m = t*B + b ----
#pragma unroll
  for (int j = 0; j < 5; ++j) {
    int col = n0g + j * 16 + lm;
    float bo = bias[col];
#pragma unroll
    for (int i = 0; i < 2; ++i) {
#pragma unroll
      for (int r = 0; r < 8; ++r) {
        int m = wgM + i * 16 + r + 8 * half;
        int t = m >> 7, b = m & (B - 1);
        out[((size_t)b * T + t) * V + col] = acc[i][j][r] + bo;
      }
    }
  }
}

// ================= small helper kernels =================
__global__ void init_state(const float* __restrict__ h0, const float* __restrict__ c0,
                           float* __restrict__ h, float* __restrict__ c,
                           const float* __restrict__ b_ih, const float* __restrict__ b_hh,
                           float* __restrict__ bsum)
{
  int i = blockIdx.x * blockDim.x + threadIdx.x;
  if (i < B * H) { h[i] = h0[i]; c[i] = c0[i]; }
  if (i < H4)    bsum[i] = b_ih[i] + b_hh[i];
}

__global__ void wout_to_bf16_t(const float* __restrict__ W, __bf16* __restrict__ Wt)
{
  int idx = blockIdx.x * blockDim.x + threadIdx.x;   // over V * 2H
  if (idx >= V * H2) return;
  int v = idx % V, k = idx / V;                      // coalesced read of W[k][v]
  Wt[(size_t)v * H2 + k] = (__bf16)W[(size_t)k * V + v];
}

__global__ void lstm_pointwise(const float* __restrict__ gates,
                               float* __restrict__ h, float* __restrict__ c,
                               __bf16* __restrict__ hcat_t)
{
  int i = blockIdx.x * blockDim.x + threadIdx.x;
  if (i >= B * H) return;
  int b = i >> 9, j = i & (H - 1);                   // H == 512
  const float* g = gates + (size_t)b * H4;
  float ig = g[j], fg = g[H + j], gg = g[2 * H + j], og = g[3 * H + j];
  float cn = sigf(fg) * c[i] + sigf(ig) * tanhf(gg);
  float hn = sigf(og) * tanhf(cn);
  c[i] = cn; h[i] = hn;
  hcat_t[(size_t)b * H2 + j] = (__bf16)hn;           // first half of [h, attended]
}

// one block (256 threads = 8 waves) per batch element
__global__ void __launch_bounds__(256)
attention_step(const float* __restrict__ Vproj, const float* __restrict__ hW,
               const float* __restrict__ wo, const float* __restrict__ area,
               __bf16* __restrict__ hcat_t)
{
  int b = blockIdx.x;
  __shared__ float sScores[F];
  __shared__ float sAlpha[F];
  int tid = threadIdx.x, lane = tid & 31, wv = tid >> 5;
  const float* hw = hW + (size_t)b * H;

  // scores[f] = sum_h tanh(Vproj[b,f,h] + hW[b,h]) * wo[h]
  for (int f = wv; f < F; f += 8) {
    const float* vp = Vproj + ((size_t)b * F + f) * H;
    float s = 0.f;
    for (int hh = lane; hh < H; hh += 32)
      s += tanhf(vp[hh] + hw[hh]) * wo[hh];
    for (int m = 16; m > 0; m >>= 1) s += __shfl_xor(s, m, 32);
    if (lane == 0) sScores[f] = s;
  }
  __syncthreads();

  // softmax over F=49 in wave 0
  if (wv == 0) {
    float s0 = (lane < F)      ? sScores[lane]      : -3.0e38f;
    float s1 = (lane + 32 < F) ? sScores[lane + 32] : -3.0e38f;
    float mx = fmaxf(s0, s1);
    for (int m = 16; m > 0; m >>= 1) mx = fmaxf(mx, __shfl_xor(mx, m, 32));
    float e0 = (lane < F)      ? expf(s0 - mx) : 0.f;
    float e1 = (lane + 32 < F) ? expf(s1 - mx) : 0.f;
    float sum = e0 + e1;
    for (int m = 16; m > 0; m >>= 1) sum += __shfl_xor(sum, m, 32);
    float inv = 1.f / sum;
    if (lane < F)      sAlpha[lane]      = e0 * inv;
    if (lane + 32 < F) sAlpha[lane + 32] = e1 * inv;
  }
  __syncthreads();

  // attended[h] = sum_f area[b,h,f] * alpha[f]   (area contiguous in f)
  for (int hh = tid; hh < H; hh += 256) {
    const float* ar = area + ((size_t)b * H + hh) * F;
    float acc = 0.f;
#pragma unroll
    for (int f = 0; f < F; ++f) acc += ar[f] * sAlpha[f];
    hcat_t[(size_t)b * H2 + H + hh] = (__bf16)acc;   // second half of [h, attended]
  }
}

// ================= host-side orchestration =================
static inline int cdiv(int a, int b) { return (a + b - 1) / b; }

extern "C" void kernel_launch(void* const* d_in, const int* in_sizes, int n_in,
                              void* d_out, int out_size, void* d_ws, size_t ws_size,
                              hipStream_t stream)
{
  (void)in_sizes; (void)n_in; (void)out_size; (void)ws_size;
  const int*   cap   = (const int*)  d_in[0];
  const float* gfeat = (const float*)d_in[1];
  const float* area  = (const float*)d_in[2];
  const float* h0    = (const float*)d_in[3];
  const float* c0    = (const float*)d_in[4];
  const float* emb   = (const float*)d_in[5];
  const float* W_ih  = (const float*)d_in[6];   // [2H, 4H]
  const float* W_hh  = (const float*)d_in[7];   // [H, 4H]
  const float* b_ih  = (const float*)d_in[8];
  const float* b_hh  = (const float*)d_in[9];
  const float* Wv    = (const float*)d_in[10];  // [H, H]
  const float* Wh    = (const float*)d_in[11];  // [H, H]
  const float* wo    = (const float*)d_in[12];  // [H]
  const float* W_out = (const float*)d_in[13];  // [2H, V]
  const float* b_out = (const float*)d_in[14];  // [V]
  float* out = (float*)d_out;                   // [B, T, V]

  // --- workspace carve (all write-before-read except h/c which we init) ---
  char* wsp = (char*)d_ws;
  auto carve = [&](size_t bytes) { char* p = wsp; wsp += (bytes + 255) & ~(size_t)255; return p; };
  float*  Xproj  = (float*) carve(sizeof(float) * (size_t)T * B * H4);   // [T*B, 4H]
  float*  Gbias  = (float*) carve(sizeof(float) * (size_t)B * H4);       // [B, 4H]
  float*  bsum   = (float*) carve(sizeof(float) * H4);
  float*  Vproj  = (float*) carve(sizeof(float) * (size_t)B * F * H);    // [B*F, H]
  float*  gates  = (float*) carve(sizeof(float) * (size_t)B * H4);
  float*  hbuf   = (float*) carve(sizeof(float) * (size_t)B * H);
  float*  cbuf   = (float*) carve(sizeof(float) * (size_t)B * H);
  float*  hWbuf  = (float*) carve(sizeof(float) * (size_t)B * H);
  __bf16* HcatBF = (__bf16*)carve(sizeof(__bf16) * (size_t)T * B * H2);  // [T*B, 2H]
  __bf16* WoutT  = (__bf16*)carve(sizeof(__bf16) * (size_t)V * H2);      // [V, 2H]

  // --- phase 0: init + weight conversion ---
  init_state<<<cdiv(B * H, 256), 256, 0, stream>>>(h0, c0, hbuf, cbuf, b_ih, b_hh, bsum);
  wout_to_bf16_t<<<cdiv(V * H2, 256), 256, 0, stream>>>(W_out, WoutT);

  // --- phase 1: time-invariant GEMMs (f32 WMMA) ---
  {// Gbias = global @ W_ih[H:2H,:] + (b_ih+b_hh)   M=128, N=2048, K=512
    int Nt = H4 / 16, tiles = (B / 16) * Nt;
    gemm_f32_wmma<ARow><<<cdiv(tiles * 32, 256), 256, 0, stream>>>(
        ARow{gfeat, H}, W_ih + (size_t)H * H4, H4, Gbias, H4, Nt, tiles, H,
        nullptr, nullptr, bsum);
  }
  {// Xproj[t*B+b,:] = embedding[tok] @ W_ih[0:H,:]  M=2560, N=2048, K=512
    int Nt = H4 / 16, tiles = (T * B / 16) * Nt;
    gemm_f32_wmma<AEmb><<<cdiv(tiles * 32, 256), 256, 0, stream>>>(
        AEmb{emb, cap}, W_ih, H4, Xproj, H4, Nt, tiles, H,
        nullptr, nullptr, nullptr);
  }
  {// Vproj = feat @ Wv   M=B*F=6272, N=512, K=512
    int Nt = H / 16, tiles = (B * F / 16) * Nt;
    gemm_f32_wmma<AArea><<<cdiv(tiles * 32, 256), 256, 0, stream>>>(
        AArea{area}, Wv, H, Vproj, H, Nt, tiles, H,
        nullptr, nullptr, nullptr);
  }

  // --- phase 2: sequential recurrence (f32 WMMA for the small GEMMs) ---
  for (int t = 0; t < T; ++t) {
    __bf16* hcat_t = HcatBF + (size_t)t * B * H2;
    {// gates = h @ W_hh + Xproj[t] + Gbias     M=128, N=2048, K=512
      int Nt = H4 / 16, tiles = (B / 16) * Nt;
      gemm_f32_wmma<ARow><<<cdiv(tiles * 32, 256), 256, 0, stream>>>(
          ARow{hbuf, H}, W_hh, H4, gates, H4, Nt, tiles, H,
          Xproj + (size_t)t * B * H4, Gbias, nullptr);
    }
    lstm_pointwise<<<cdiv(B * H, 256), 256, 0, stream>>>(gates, hbuf, cbuf, hcat_t);
    {// hW = h_new @ Wh                          M=128, N=512, K=512
      int Nt = H / 16, tiles = (B / 16) * Nt;
      gemm_f32_wmma<ARow><<<cdiv(tiles * 32, 256), 256, 0, stream>>>(
          ARow{hbuf, H}, Wh, H, hWbuf, H, Nt, tiles, H,
          nullptr, nullptr, nullptr);
    }
    attention_step<<<B, 256, 0, stream>>>(Vproj, hWbuf, wo, area, hcat_t);
  }

  // --- phase 3: one big BF16 WMMA GEMM for all logits ---
  // grid: (2560/256 M-blocks) x (10000/80 N-supers) = 10 x 125 = 1250 full WGs
  outproj_bf16_wmma<<<1250, 256, 0, stream>>>(HcatBF, WoutT, b_out, out);
}
